// CausalSelfAttention_25847113187471
// MI455X (gfx1250) — matrix operand order repaired and checked by reading, not compiled, not run on hardware
//
#include <hip/hip_runtime.h>
#include <hip/hip_bf16.h>
#include <math.h>

// ---------------------------------------------------------------------------
// Problem constants (from the reference)
// ---------------------------------------------------------------------------
constexpr int Bb   = 2;
constexpr int T    = 4096;
constexpr int C    = 1024;
constexpr int H    = 16;
constexpr int N    = 64;          // head dim
constexpr int L    = 256;         // chunk length
constexpr int NB   = 2;
constexpr int NC   = T / L;       // 16 chunks
constexpr int NSO  = NC - NB - 1; // 13 scan/outer chunks
constexpr int TBF  = (NB + 1) * L;// 768 front-block rows / gathered keys
constexpr int LORA = 32;
constexpr int BT   = Bb * T;      // 8192 rows
constexpr int BTC  = BT * C;      // 8,388,608
constexpr float SCALE = 0.125f;   // N^-0.5

// ---------------------------------------------------------------------------
// WMMA vector types (probe-confirmed builtin signature)
// ---------------------------------------------------------------------------
typedef _Float16 v16h __attribute__((ext_vector_type(16)));
typedef _Float16 v8h  __attribute__((ext_vector_type(8)));
typedef float    v8f  __attribute__((ext_vector_type(8)));

union V16 { v16h v; v8h h[2]; };

__device__ __forceinline__ v8f wmma_f16(v16h a, v16h b, v8f c) {
  return __builtin_amdgcn_wmma_f32_16x16x32_f16(false, a, false, b, (short)0, c,
                                                false, false);
}

__device__ __forceinline__ float wred_add32(float v) {
  #pragma unroll
  for (int m = 1; m < 32; m <<= 1) v += __shfl_xor(v, m, 32);
  return v;
}
__device__ __forceinline__ float wred_add16(float v) {
  #pragma unroll
  for (int m = 1; m < 16; m <<= 1) v += __shfl_xor(v, m, 32);
  return v;
}
__device__ __forceinline__ float wred_max16(float v) {
  #pragma unroll
  for (int m = 1; m < 16; m <<= 1) v = fmaxf(v, __shfl_xor(v, m, 32));
  return v;
}

// ---------------------------------------------------------------------------
// f32 -> f16 pack
// ---------------------------------------------------------------------------
__global__ void __launch_bounds__(256)
k_f32_to_f16(const float* __restrict__ in, _Float16* __restrict__ out, int n) {
  int i = blockIdx.x * 256 + threadIdx.x;
  if (i < n) out[i] = (_Float16)in[i];
}

// ---------------------------------------------------------------------------
// WMMA GEMM: out[M,Nn] = X[M,K] * W[Nn,K]^T (+ res).  One wave = 16x64 tile.
// A-frag: lane(half,m) reads X[m, k0+16*half+0..15]  (contiguous 32B)
// B-frag: lane(half,n) reads W[n, k0+16*half+0..15]  (contiguous 32B)
// Ping-pong double buffering (unroll-by-2, no register rotation copies):
// WMMAs for step k consume fragments loaded one k-step earlier while the
// other buffer's loads are still in flight, so LOADcnt waits are partial.
// K must be a multiple of 64.
// ---------------------------------------------------------------------------
__global__ void __launch_bounds__(128)
k_gemm_f16(const _Float16* __restrict__ X, const _Float16* __restrict__ W,
           const float* __restrict__ res, float* __restrict__ out,
           int M, int Nn, int K) {
  const int lane = threadIdx.x & 31;
  const int wave = threadIdx.x >> 5;
  const int half = lane >> 4;
  const int l16  = lane & 15;
  const int m0 = blockIdx.x * 16;
  const int n0 = blockIdx.y * 256 + wave * 64;

  const _Float16* xrow = X + (size_t)(m0 + l16) * K + 16 * half;
  const _Float16* wr[4];
  #pragma unroll
  for (int j = 0; j < 4; ++j)
    wr[j] = W + (size_t)(n0 + j * 16 + l16) * K + 16 * half;

  auto ldA = [&](V16& a, int k0) {
    a.h[0] = *(const v8h*)(xrow + k0);
    a.h[1] = *(const v8h*)(xrow + k0 + 8);
  };
  auto ldB = [&](V16& b, const _Float16* w, int k0) {
    b.h[0] = *(const v8h*)(w + k0);
    b.h[1] = *(const v8h*)(w + k0 + 8);
  };

  v8f acc[4] = {};
  V16 a0, b0[4], a1, b1[4];

  ldA(a0, 0);
  #pragma unroll
  for (int j = 0; j < 4; ++j) ldB(b0[j], wr[j], 0);

  for (int k0 = 0; k0 < K; k0 += 64) {
    // prefetch buffer 1 @ k0+32 (always in range: K % 64 == 0)
    ldA(a1, k0 + 32);
    #pragma unroll
    for (int j = 0; j < 4; ++j) ldB(b1[j], wr[j], k0 + 32);
    // consume buffer 0 (loaded one k-step ago)
    #pragma unroll
    for (int j = 0; j < 4; ++j) acc[j] = wmma_f16(a0.v, b0[j].v, acc[j]);
    // prefetch buffer 0 @ k0+64
    if (k0 + 64 < K) {
      ldA(a0, k0 + 64);
      #pragma unroll
      for (int j = 0; j < 4; ++j) ldB(b0[j], wr[j], k0 + 64);
    }
    // consume buffer 1
    #pragma unroll
    for (int j = 0; j < 4; ++j) acc[j] = wmma_f16(a1.v, b1[j].v, acc[j]);
  }

  #pragma unroll
  for (int j = 0; j < 4; ++j) {
    #pragma unroll
    for (int r = 0; r < 8; ++r) {
      int row = m0 + r + 8 * half;
      int col = n0 + j * 16 + l16;
      float v = acc[j][r];
      if (res) v += res[(size_t)row * Nn + col];
      out[(size_t)row * Nn + col] = v;
    }
  }
}

// ---------------------------------------------------------------------------
// LoRA first stage: xv1[m,l] = sum_k x[m,k] * V1[k,l]   (l = lane)
// ---------------------------------------------------------------------------
__global__ void __launch_bounds__(256)
k_lora(const float* __restrict__ x, const float* __restrict__ V1,
       float* __restrict__ xv1) {
  int l = threadIdx.x & 31;
  int m = blockIdx.x * 8 + (threadIdx.x >> 5);
  const float* xr = x + (size_t)m * C;
  float s = 0.f;
  for (int k = 0; k < C; ++k) s = fmaf(xr[k], V1[k * LORA + l], s);
  xv1[m * LORA + l] = s;
}

// ---------------------------------------------------------------------------
// v mix: vM = vraw + (v1 - vraw) * sigmoid(v0[c] + xv1[m,:] . V2[:,c])
// ---------------------------------------------------------------------------
__global__ void __launch_bounds__(256)
k_vmix(const float* __restrict__ vraw, const float* __restrict__ v1in,
       const float* __restrict__ xv1, const float* __restrict__ V2,
       const float* __restrict__ v0, float* __restrict__ vM) {
  int idx = blockIdx.x * 256 + threadIdx.x;
  int m = idx >> 10, c = idx & 1023;
  float s = v0[c];
  const float* xl = xv1 + m * LORA;
  #pragma unroll
  for (int l = 0; l < LORA; ++l) s = fmaf(xl[l], V2[l * C + c], s);
  float sig = 1.f / (1.f + expf(-s));
  float vr = vraw[idx];
  vM[idx] = vr + (v1in[idx] - vr) * sig;
}

// ---------------------------------------------------------------------------
// kw = 1 + elu(x . Wkw[:,h]),  ld = -softplus(x . Wsd[:,h])  (wave per (bt,h))
// ---------------------------------------------------------------------------
__global__ void __launch_bounds__(256)
k_kwld(const float* __restrict__ x, const float* __restrict__ Wkw,
       const float* __restrict__ Wsd, float* __restrict__ kw,
       float* __restrict__ ld) {
  int lane = threadIdx.x & 31;
  int gid = blockIdx.x * 8 + (threadIdx.x >> 5);
  int h = gid & (H - 1);
  int bt = gid >> 4;
  const float* xr = x + (size_t)bt * C;
  float sw = 0.f, sd = 0.f;
  for (int k = lane; k < C; k += 32) {
    float xv = xr[k];
    sw = fmaf(xv, Wkw[k * H + h], sw);
    sd = fmaf(xv, Wsd[k * H + h], sd);
  }
  sw = wred_add32(sw);
  sd = wred_add32(sd);
  if (lane == 0) {
    kw[bt * H + h] = 1.f + (sw > 0.f ? sw : expm1f(sw));
    ld[bt * H + h] = -(fmaxf(sd, 0.f) + log1pf(expf(-fabsf(sd))));
  }
}

// ---------------------------------------------------------------------------
// Fused token-shift + per-head LayerNorm + RoPE.  Wave per (bt,h); lane = 2 n.
// Produces qB,kB (f16, roped) and kz,vB (f32).
// ---------------------------------------------------------------------------
__global__ void __launch_bounds__(256)
k_slr(const float* __restrict__ qA, const float* __restrict__ kA,
      const float* __restrict__ vM, const float* __restrict__ xs_q,
      const float* __restrict__ xs_k, const float* __restrict__ xs_v,
      const float* __restrict__ lnqg, const float* __restrict__ lnqb,
      const float* __restrict__ lnkg, const float* __restrict__ lnkb,
      const float* __restrict__ lndg, const float* __restrict__ lndb,
      const float* __restrict__ cosb, const float* __restrict__ sinb,
      _Float16* __restrict__ qB, _Float16* __restrict__ kB,
      float* __restrict__ kz, float* __restrict__ vB) {
  int lane = threadIdx.x & 31;
  int gid = blockIdx.x * 8 + (threadIdx.x >> 5);
  int h = gid & (H - 1);
  int bt = gid >> 4;
  int t = bt & (T - 1);
  int n0 = lane * 2;
  int hn = h * N + n0;
  size_t base = (size_t)bt * C + hn;
  bool sh = (t > 0);  // nosink: sink_mask is exactly (t == 0)

  auto shiftpair = [&](const float* A, const float* xs, float& e0, float& e1) {
    e0 = A[base]; e1 = A[base + 1];
    if (sh) {
      e0 += xs[hn]     * (A[base - C]     - e0);
      e1 += xs[hn + 1] * (A[base - C + 1] - e1);
    }
  };
  auto lnpair = [&](float& e0, float& e1, const float* g, const float* bi) {
    float mean = wred_add32(e0 + e1) * (1.f / 64.f);
    float d0 = e0 - mean, d1 = e1 - mean;
    float var = wred_add32(d0 * d0 + d1 * d1) * (1.f / 64.f);
    float inv = rsqrtf(var + 1e-5f);
    e0 = d0 * inv * g[n0] + bi[n0];
    e1 = d1 * inv * g[n0 + 1] + bi[n0 + 1];
  };

  float c0 = cosb[t * N + n0], c1 = cosb[t * N + n0 + 1];
  float s0 = sinb[t * N + n0], s1 = sinb[t * N + n0 + 1];

  float q0, q1; shiftpair(qA, xs_q, q0, q1); lnpair(q0, q1, lnqg, lnqb);
  float qr0 = c0 * q0 + s0 * q1, qr1 = c1 * q1 + s1 * q0;
  qB[base] = (_Float16)qr0; qB[base + 1] = (_Float16)qr1;

  float k0, k1; shiftpair(kA, xs_k, k0, k1); lnpair(k0, k1, lnkg, lnkb);
  float kr0 = c0 * k0 + s0 * k1, kr1 = c1 * k1 + s1 * k0;
  kB[base] = (_Float16)kr0; kB[base + 1] = (_Float16)kr1;
  // kz = LN( rope(k_roped, cos, -sin) )
  float z0 = c0 * kr0 - s0 * kr1, z1 = c1 * kr1 - s1 * kr0;
  lnpair(z0, z1, lndg, lndb);
  kz[base] = z0; kz[base + 1] = z1;

  float ve0, ve1; shiftpair(vM, xs_v, ve0, ve1);
  vB[base] = ve0; vB[base + 1] = ve1;
}

// ---------------------------------------------------------------------------
// Sequential state scan over 13 chunks; wave per (b,h,row l); lane = 2 n.
// ---------------------------------------------------------------------------
__global__ void __launch_bounds__(256)
k_scan(const float* __restrict__ kz, const float* __restrict__ vB,
       const float* __restrict__ kw, const float* __restrict__ ld,
       float* __restrict__ sks, float* __restrict__ svs) {
  int lane = threadIdx.x & 31;
  int gid = blockIdx.x * 8 + (threadIdx.x >> 5);
  int l = gid & (L - 1);
  int h = (gid >> 8) & (H - 1);
  int b = gid >> 12;
  int n0 = lane * 2;

  size_t o0 = (size_t)(b * T + l) * C + h * N + n0;  // chunk 0
  float sk0 = kz[o0], sk1 = kz[o0 + 1];
  float sv0 = vB[o0], sv1 = vB[o0 + 1];
  float svlen = sqrtf(wred_add32(sv0 * sv0 + sv1 * sv1));

  for (int i = 0; i < NSO; ++i) {
    int t = (i + 1) * L + l;
    size_t o = (size_t)(b * T + t) * C + h * N + n0;
    float d = expf(ld[(b * T + t) * H + h]);
    float w = kw[(b * T + t) * H + h];
    float om = 1.f - d;
    sk0 = sk0 * d + om * w * kz[o];
    sk1 = sk1 * d + om * w * kz[o + 1];
    sv0 = sv0 * d + om * vB[o];
    sv1 = sv1 * d + om * vB[o + 1];
    float nrm = sqrtf(wred_add32(sv0 * sv0 + sv1 * sv1));
    float f = svlen / (nrm + 1e-6f);
    sv0 *= f; sv1 *= f;
    size_t so = ((((size_t)(b * H + h) * NSO + i) * L + l) * N) + n0;
    sks[so] = sk0; sks[so + 1] = sk1;
    svs[so] = sv0; svs[so + 1] = sv1;
  }
}

// ---------------------------------------------------------------------------
// Gather per-(b,h,chunk) attention K (row-major key x dim, f16, temp-scaled)
// and V^T (dim x key, f16) so both WMMA B-fragments are contiguous loads.
// ---------------------------------------------------------------------------
__global__ void __launch_bounds__(256)
k_gather(const _Float16* __restrict__ kB, const float* __restrict__ vB,
         const float* __restrict__ sks, const float* __restrict__ svs,
         const float* __restrict__ ft, const float* __restrict__ st,
         _Float16* __restrict__ KG, _Float16* __restrict__ VGT) {
  int idx = blockIdx.x * 256 + threadIdx.x;
  int n = idx & 63;
  int j = (idx % (TBF * N)) >> 6;
  int rest = idx / (TBF * N);
  int c = rest & 15;
  int h = (rest >> 4) & 15;
  int b = rest >> 8;

  float kval, vval;
  if (c <= NB) {                         // front block: keys = k[0:768]*ft
    size_t o = (size_t)(b * T + j) * C + h * N + n;
    kval = (float)kB[o] * ft[h];
    vval = vB[o];
  } else {
    int i = c - (NB + 1);
    if (j < L) {                         // recurrent state, scaled by st
      size_t so = ((((size_t)(b * H + h) * NSO + i) * L + j) * N) + n;
      kval = sks[so] * st[h];
      vval = svs[so];
    } else {                             // sliding window chunks c-1, c
      int t = (j < 2 * L) ? (c - 1) * L + (j - L) : c * L + (j - 2 * L);
      size_t o = (size_t)(b * T + t) * C + h * N + n;
      kval = (float)kB[o] * ft[h];
      vval = vB[o];
    }
  }
  size_t cb = (size_t)((b * H + h) * NC + c);
  KG [cb * TBF * N + (size_t)j * N + n] = (_Float16)kval;
  VGT[cb * N * TBF + (size_t)n * TBF + j] = (_Float16)vval;
}

// ---------------------------------------------------------------------------
// Attention: wave owns 16 query rows of one (b,h,chunk); flash-style online
// softmax over 32-key tiles.  Scores via 4 WMMAs; V fragments are fetched
// immediately after (independent of softmax) so their latency overlaps the
// softmax VALU work and the LDS P round-trip; P routed C/D->A via LDS;
// PV via 4 WMMAs into 16x64 f32 accumulators.
// ---------------------------------------------------------------------------
__global__ void __launch_bounds__(256)
k_attn(const _Float16* __restrict__ qB, const _Float16* __restrict__ KG,
       const _Float16* __restrict__ VGT, float* __restrict__ y) {
  __shared__ alignas(16) _Float16 plds[8][16 * 32];
  const int lane = threadIdx.x & 31;
  const int wave = threadIdx.x >> 5;
  const int half = lane >> 4;
  const int l16  = lane & 15;

  int blk = blockIdx.x;                  // [0, B*H*NC*2)
  int qtile = (blk & 1) * 8 + wave;      // 16 query tiles per chunk
  int c = (blk >> 1) & 15;
  int h = (blk >> 5) & 15;
  int b = blk >> 9;
  int m0 = qtile * 16;
  int off = (c <= NB) ? c * L : 2 * L;   // causal offset: key j valid if j<=off+row

  size_t cb = (size_t)((b * H + h) * NC + c);
  const _Float16* Kbase = KG  + cb * TBF * N;
  const _Float16* Vbase = VGT + cb * N * TBF;

  // Q fragments (head dim 64 = 2 K-chunks of 32)
  V16 aq0, aq1;
  {
    int t = c * L + m0 + l16;
    const _Float16* qrow = qB + (size_t)(b * T + t) * C + h * N;
    aq0.h[0] = *(const v8h*)(qrow + 16 * half);
    aq0.h[1] = *(const v8h*)(qrow + 8 + 16 * half);
    aq1.h[0] = *(const v8h*)(qrow + 32 + 16 * half);
    aq1.h[1] = *(const v8h*)(qrow + 40 + 16 * half);
  }

  v8f o[4] = {};
  float mrun[8], lrun[8], p0[8], p1[8];
  #pragma unroll
  for (int r = 0; r < 8; ++r) { mrun[r] = -INFINITY; lrun[r] = 0.f; }

  int kmax = off + m0 + 15;
  int ktiles = (kmax + 32) / 32;
  _Float16* lp = plds[wave];

  for (int kt = 0; kt < ktiles; ++kt) {
    // ---- scores: S(16x32) = Q(16x64) . K^T, two 16x16 output tiles -------
    v8f sc0 = {}, sc1 = {};
    {
      const _Float16* kr0 = Kbase + (size_t)(kt * 32 + l16) * N + 16 * half;
      V16 bk;
      bk.h[0] = *(const v8h*)(kr0);      bk.h[1] = *(const v8h*)(kr0 + 8);
      sc0 = wmma_f16(aq0.v, bk.v, sc0);
      bk.h[0] = *(const v8h*)(kr0 + 32); bk.h[1] = *(const v8h*)(kr0 + 40);
      sc0 = wmma_f16(aq1.v, bk.v, sc0);
      const _Float16* kr1 = kr0 + 16 * N;
      bk.h[0] = *(const v8h*)(kr1);      bk.h[1] = *(const v8h*)(kr1 + 8);
      sc1 = wmma_f16(aq0.v, bk.v, sc1);
      bk.h[0] = *(const v8h*)(kr1 + 32); bk.h[1] = *(const v8h*)(kr1 + 40);
      sc1 = wmma_f16(aq1.v, bk.v, sc1);
    }
    // ---- V fragments for this key tile (independent of softmax) ---------
    V16 bv[4];
    #pragma unroll
    for (int j = 0; j < 4; ++j) {
      const _Float16* vr = Vbase + (size_t)(j * 16 + l16) * TBF + kt * 32 + 16 * half;
      bv[j].h[0] = *(const v8h*)vr;
      bv[j].h[1] = *(const v8h*)(vr + 8);
    }
    // ---- mask + online softmax ------------------------------------------
    #pragma unroll
    for (int r = 0; r < 8; ++r) {
      int row = m0 + r + 8 * half;
      int j0 = kt * 32 + l16;
      float v0s = (j0      <= off + row) ? sc0[r] * SCALE : -INFINITY;
      float v1s = (j0 + 16 <= off + row) ? sc1[r] * SCALE : -INFINITY;
      float mx = wred_max16(fmaxf(v0s, v1s));
      float mnew = fmaxf(mrun[r], mx);
      float alpha = __expf(mrun[r] - mnew);
      float e0 = __expf(v0s - mnew);
      float e1 = __expf(v1s - mnew);
      float rs = wred_add16(e0 + e1);
      lrun[r] = lrun[r] * alpha + rs;
      mrun[r] = mnew;
      #pragma unroll
      for (int j = 0; j < 4; ++j) o[j][r] *= alpha;
      p0[r] = e0; p1[r] = e1;
    }
    // ---- P: C/D layout -> A fragment via per-wave LDS (16x32 f16) --------
    #pragma unroll
    for (int r = 0; r < 8; ++r) {
      int row = r + 8 * half;
      lp[row * 32 + l16]      = (_Float16)p0[r];
      lp[row * 32 + 16 + l16] = (_Float16)p1[r];
    }
    asm volatile("s_wait_dscnt 0" ::: "memory");
    V16 pa;
    pa.h[0] = *(const v8h*)(lp + l16 * 32 + 16 * half);
    pa.h[1] = *(const v8h*)(lp + l16 * 32 + 8 + 16 * half);
    // ---- PV: O(16x64) += P(16x32) . V(32x64) -----------------------------
    #pragma unroll
    for (int j = 0; j < 4; ++j) o[j] = wmma_f16(pa.v, bv[j].v, o[j]);
  }

  // ---- normalize and store y (B,T,C) f32 ---------------------------------
  #pragma unroll
  for (int r = 0; r < 8; ++r) {
    float inv = 1.f / lrun[r];
    int t = c * L + m0 + r + 8 * half;
    size_t yb = (size_t)(b * T + t) * C + h * N;
    #pragma unroll
    for (int j = 0; j < 4; ++j) y[yb + j * 16 + l16] = o[j][r] * inv;
  }
}

// ---------------------------------------------------------------------------
// Host-side launch.  Workspace layout (bytes, all 256B-aligned; peak ~302 MB):
//   [xh 16M][Wq/Wk/Wv/Wp f16 4x2M][qA 32M][kA 32M][vA 32M][xv1 1M][vM 32M]
//   [qB 16M][kB 16M][kz 32M][vB 32M][kw .5M][ld .5M][sks 26M][svs 26M]
//   reuse: KG/VGT over qA..vA (96M), y over xv1+vM (33M), yh over xh (16M)
// ---------------------------------------------------------------------------
extern "C" void kernel_launch(void* const* d_in, const int* in_sizes, int n_in,
                              void* d_out, int out_size, void* d_ws, size_t ws_size,
                              hipStream_t stream) {
  const float* residual = (const float*)d_in[0];
  const float* x        = (const float*)d_in[1];
  const float* v1in     = (const float*)d_in[2];
  /* d_in[3] sink_mask: equals (t==0); handled analytically */
  const float* cosb = (const float*)d_in[4];
  const float* sinb = (const float*)d_in[5];
  const float* Wq   = (const float*)d_in[6];
  const float* Wk   = (const float*)d_in[7];
  const float* Wv   = (const float*)d_in[8];
  const float* Wp   = (const float*)d_in[9];
  const float* xs_q = (const float*)d_in[10];
  const float* xs_k = (const float*)d_in[11];
  const float* xs_v = (const float*)d_in[12];
  const float* lnqg = (const float*)d_in[13];
  const float* lnqb = (const float*)d_in[14];
  const float* lnkg = (const float*)d_in[15];
  const float* lnkb = (const float*)d_in[16];
  const float* lndg = (const float*)d_in[17];
  const float* lndb = (const float*)d_in[18];
  const float* v0c  = (const float*)d_in[19];
  const float* V1   = (const float*)d_in[20];
  const float* V2   = (const float*)d_in[21];
  const float* Wsd  = (const float*)d_in[22];
  const float* Wkw  = (const float*)d_in[23];
  const float* ftp  = (const float*)d_in[24];
  const float* stp  = (const float*)d_in[25];
  float* out = (float*)d_out;
  char* ws = (char*)d_ws;

  constexpr size_t SZ_XH  = (size_t)BTC * 2;
  constexpr size_t SZ_WH  = (size_t)C * C * 2;
  constexpr size_t SZ_F32 = (size_t)BTC * 4;
  constexpr size_t SZ_XV1 = (size_t)BT * LORA * 4;
  constexpr size_t SZ_H16 = (size_t)BTC * 2;
  constexpr size_t SZ_KW  = (size_t)BT * H * 4;
  constexpr size_t SZ_SK  = (size_t)Bb * H * NSO * L * N * 4;
  constexpr size_t SZ_KG  = (size_t)Bb * H * NC * TBF * N * 2;

  constexpr size_t O_XH  = 0;
  constexpr size_t O_WQH = O_XH + SZ_XH;
  constexpr size_t O_WKH = O_WQH + SZ_WH;
  constexpr size_t O_WVH = O_WKH + SZ_WH;
  constexpr size_t O_WPH = O_WVH + SZ_WH;
  constexpr size_t O_QA  = O_WPH + SZ_WH;
  constexpr size_t O_KA  = O_QA + SZ_F32;
  constexpr size_t O_VA  = O_KA + SZ_F32;
  constexpr size_t O_XV1 = O_VA + SZ_F32;
  constexpr size_t O_VM  = O_XV1 + SZ_XV1;
  constexpr size_t O_QB  = O_VM + SZ_F32;
  constexpr size_t O_KB  = O_QB + SZ_H16;
  constexpr size_t O_KZ  = O_KB + SZ_H16;
  constexpr size_t O_VB  = O_KZ + SZ_F32;
  constexpr size_t O_KW  = O_VB + SZ_F32;
  constexpr size_t O_LD  = O_KW + SZ_KW;
  constexpr size_t O_SKS = O_LD + SZ_KW;
  constexpr size_t O_SVS = O_SKS + SZ_SK;
  // reuse (dead after the fused elementwise stage)
  constexpr size_t O_KG  = O_QA;             // 48M over qA+kA
  constexpr size_t O_VGT = O_KG + SZ_KG;     // 48M over kA+vA
  constexpr size_t O_Y   = O_XV1;            // 32M over xv1+vM
  constexpr size_t O_YH  = O_XH;             // 16M over xh

  _Float16* xh  = (_Float16*)(ws + O_XH);
  _Float16* wqh = (_Float16*)(ws + O_WQH);
  _Float16* wkh = (_Float16*)(ws + O_WKH);
  _Float16* wvh = (_Float16*)(ws + O_WVH);
  _Float16* wph = (_Float16*)(ws + O_WPH);
  float* qA  = (float*)(ws + O_QA);
  float* kA  = (float*)(ws + O_KA);
  float* vA  = (float*)(ws + O_VA);
  float* xv1 = (float*)(ws + O_XV1);
  float* vM  = (float*)(ws + O_VM);
  _Float16* qBf = (_Float16*)(ws + O_QB);
  _Float16* kBf = (_Float16*)(ws + O_KB);
  float* kz  = (float*)(ws + O_KZ);
  float* vB  = (float*)(ws + O_VB);
  float* kw  = (float*)(ws + O_KW);
  float* ld  = (float*)(ws + O_LD);
  float* sks = (float*)(ws + O_SKS);
  float* svs = (float*)(ws + O_SVS);
  _Float16* KG  = (_Float16*)(ws + O_KG);
  _Float16* VGT = (_Float16*)(ws + O_VGT);
  float* yb = (float*)(ws + O_Y);
  _Float16* yh = (_Float16*)(ws + O_YH);

  // 1) pack activations + weights to f16
  k_f32_to_f16<<<BTC / 256, 256, 0, stream>>>(x, xh, BTC);
  k_f32_to_f16<<<(C * C) / 256, 256, 0, stream>>>(Wq, wqh, C * C);
  k_f32_to_f16<<<(C * C) / 256, 256, 0, stream>>>(Wk, wkh, C * C);
  k_f32_to_f16<<<(C * C) / 256, 256, 0, stream>>>(Wv, wvh, C * C);
  k_f32_to_f16<<<(C * C) / 256, 256, 0, stream>>>(Wp, wph, C * C);

  // 2) QKV projections (WMMA)
  dim3 gg(BT / 16, C / 256);
  k_gemm_f16<<<gg, 128, 0, stream>>>(xh, wqh, nullptr, qA, BT, C, C);
  k_gemm_f16<<<gg, 128, 0, stream>>>(xh, wkh, nullptr, kA, BT, C, C);
  k_gemm_f16<<<gg, 128, 0, stream>>>(xh, wvh, nullptr, vA, BT, C, C);

  // 3) LoRA gate + v mix + head gates
  k_lora<<<BT / 8, 256, 0, stream>>>(x, V1, xv1);
  k_vmix<<<BTC / 256, 256, 0, stream>>>(vA, v1in, xv1, V2, v0c, vM);
  k_kwld<<<BT * H / 8, 256, 0, stream>>>(x, Wkw, Wsd, kw, ld);

  // 4) token shift + LN + RoPE (+ kz branch)
  k_slr<<<BT * H / 8, 256, 0, stream>>>(qA, kA, vM, xs_q, xs_k, xs_v,
                                        lnqg, lnqb, lnkg, lnkb, lndg, lndb,
                                        cosb, sinb, qBf, kBf, kz, vB);

  // 5) recurrent state scan
  k_scan<<<Bb * H * L / 8, 256, 0, stream>>>(kz, vB, kw, ld, sks, svs);

  // 6) gather per-chunk K / V^T (f16, temperature-scaled)
  int ngath = Bb * H * NC * TBF * N;
  k_gather<<<ngath / 256, 256, 0, stream>>>(kBf, vB, sks, svs, ftp, stp, KG, VGT);

  // 7) attention (front block + state/window chunks, unified mask)
  k_attn<<<Bb * H * NC * 2, 256, 0, stream>>>(qBf, KG, VGT, yb);

  // 8) output projection + residual
  k_f32_to_f16<<<BTC / 256, 256, 0, stream>>>(yb, yh, BTC);
  k_gemm_f16<<<gg, 128, 0, stream>>>(yh, wph, residual, out, BT, C, C);

  (void)in_sizes; (void)n_in; (void)out_size; (void)ws_size;
}
